// DecoderActor_53884659696277
// MI455X (gfx1250) — matrix operand order repaired and checked by reading, not compiled, not run on hardware
//
#include <hip/hip_runtime.h>
#include <hip/hip_bf16.h>

// ---------------------------------------------------------------------------
// MI455X (gfx1250) wave32 WMMA implementation of the DecoderActor reference.
//   x:[16384,20,128] f32 -> p:[16384,20,20] f32
// One wave per batch item; all GEMMs via v_wmma_f32_16x16x32_f16.
// Round 2: 2 LDS buffers/wave (Q reuses H) -> 2 blocks/WGP occupancy;
//          B-tiles preloaded per n-tile; K/Q GEMMs fused (4 indep accums).
// ---------------------------------------------------------------------------

typedef __attribute__((ext_vector_type(16))) _Float16 v16h;
typedef __attribute__((ext_vector_type(8)))  _Float16 v8h;
typedef __attribute__((ext_vector_type(8)))  float    v8f;

#define DM     128
#define NNODE  20
#define BATCH  16384
#define WAVES  8              // waves per block
#define LROW   136            // halves per LDS tile row (272B = 68 banks, conflict-free)
#define TILEH  (32 * LROW)    // halves per 32-row LDS tile buffer

__device__ __forceinline__ v8f wmma16(v16h a, v16h b, v8f c) {
    return __builtin_amdgcn_wmma_f32_16x16x32_f16(
        /*neg_a=*/false, a, /*neg_b=*/false, b,
        /*c_mod=*/(short)0, c, /*reuse_a=*/false, /*reuse_b=*/false);
}

__device__ __forceinline__ v16h cat8(v8h lo, v8h hi) {
    return __builtin_shufflevector(lo, hi, 0,1,2,3,4,5,6,7,8,9,10,11,12,13,14,15);
}

__device__ __forceinline__ v16h zero16h() {
    v16h z;
#pragma unroll
    for (int i = 0; i < 16; ++i) z[i] = (_Float16)0.f;
    return z;
}

// f32 global -> f16 A-operand chunk pair (ISA 16-bit A layout: K=kc..kc+7, kc+16..kc+23)
__device__ __forceinline__ v16h cvtA(const float* __restrict__ p) {
    v16h r;
#pragma unroll
    for (int i = 0; i < 8; ++i) { r[i] = (_Float16)p[i]; r[8 + i] = (_Float16)p[16 + i]; }
    return r;
}

// A operand (16-bit, 16x32 tile) from LDS, row-major with LROW stride.
__device__ __forceinline__ v16h ld_a_lds(const _Float16* buf, int mtile, int ks, int lane) {
    int row = mtile * 16 + (lane & 15);
    int kc  = ks * 32 + ((lane >> 4) << 3);
    const _Float16* p = buf + row * LROW + kc;
    return cat8(*(const v8h*)p, *(const v8h*)(p + 16));
}

// B operand (16-bit, 32x16 tile): lane n holds K=kc..kc+7, kc+16..kc+23 (kc = 8*(lane/16)), matching A.
__device__ __forceinline__ v16h ld_b_lds(const _Float16* buf, int ntile, int ks, int lane) {
    int row = ntile * 16 + (lane & 15);
    int kc  = ks * 32 + ((lane >> 4) << 3);
    const _Float16* p = buf + row * LROW + kc;
    return cat8(*(const v8h*)p, *(const v8h*)(p + 16));
}

// B operand from the f16 weight copy in global scratch (row-major 128x128).
__device__ __forceinline__ v16h ld_b_w(const _Float16* __restrict__ W, int ntile, int ks, int lane) {
    int row = ntile * 16 + (lane & 15);
    int kc  = ks * 32 + ((lane >> 4) << 3);
    const _Float16* p = W + row * DM + kc;
    return cat8(*(const v8h*)p, *(const v8h*)(p + 16));
}

// --------------------------- weight f32 -> f16 prep ------------------------
__global__ void cvt_weights_kernel(const float* __restrict__ Wg, const float* __restrict__ Wn,
                                   const float* __restrict__ Wk, const float* __restrict__ Wq,
                                   _Float16* __restrict__ ws) {
    int i = blockIdx.x * blockDim.x + threadIdx.x;
    if (i < DM * DM) {
        ws[i]               = (_Float16)Wn[i];
        ws[DM * DM + i]     = (_Float16)Wg[i];
        ws[2 * DM * DM + i] = (_Float16)Wk[i];
        ws[3 * DM * DM + i] = (_Float16)Wq[i];
    }
}

// ------------------------------- main kernel -------------------------------
__global__ __launch_bounds__(WAVES * 32) void decoder_actor_kernel(
    const float* __restrict__ x,
    const float* __restrict__ bg, const float* __restrict__ bn,
    const float* __restrict__ bk, const float* __restrict__ bq,
    const _Float16* __restrict__ ws, float* __restrict__ out) {

    extern __shared__ _Float16 smem[];
    const int lane = threadIdx.x & 31;
    const int wid  = threadIdx.x >> 5;
    const int item = blockIdx.x * WAVES + wid;

    // Two buffers per wave: bufH holds h, then is reused for Q. bufK holds K.
    _Float16* bufH = smem + wid * (2 * TILEH);
    _Float16* bufK = bufH + TILEH;

    const _Float16* Wn16 = ws;
    const _Float16* Wg16 = ws + DM * DM;
    const _Float16* Wk16 = ws + 2 * DM * DM;
    const _Float16* Wq16 = ws + 3 * DM * DM;

    const float* xb = x + (size_t)item * (NNODE * DM);

    const int ncol = lane & 15;
    const int rhi  = (lane >> 4) << 3;          // +8 row offset for upper lane half

    // ---- Step 1: x_graph = max over 20 node rows (lane owns 4 columns) ----
    {
        int c0 = lane * 4;
        float m0 = -1e30f, m1 = -1e30f, m2 = -1e30f, m3 = -1e30f;
#pragma unroll
        for (int r = 0; r < NNODE; ++r) {
            const float4 v = *(const float4*)(xb + r * DM + c0);
            m0 = fmaxf(m0, v.x); m1 = fmaxf(m1, v.y);
            m2 = fmaxf(m2, v.z); m3 = fmaxf(m3, v.w);
        }
        _Float16* xgt = bufK;                   // temp: 128 halves, overwritten later by K
        xgt[c0]     = (_Float16)m0; xgt[c0 + 1] = (_Float16)m1;
        xgt[c0 + 2] = (_Float16)m2; xgt[c0 + 3] = (_Float16)m3;
    }

    // ---- Step 2: build A tiles of padded x (rows 20 & 28 = x_graph) ----
    v16h a0[4], a1[4];
    {
        const int r15 = lane & 15;
#pragma unroll
        for (int ks = 0; ks < 4; ++ks) {
            int kc = ks * 32 + ((lane >> 4) << 3);
            a0[ks] = cvtA(xb + r15 * DM + kc);
            if (r15 < 4) {
                a1[ks] = cvtA(xb + (16 + r15) * DM + kc);
            } else if (r15 == 4 || r15 == 12) {       // rows 20 / 28 <- x_graph
                const _Float16* p = bufK + kc;        // xg temp
                a1[ks] = cat8(*(const v8h*)p, *(const v8h*)(p + 16));
            } else {
                a1[ks] = zero16h();
            }
        }
    }

    // Hoisted per-lane bias values for all 8 n-tiles.
    float bsum[8], bkv[8], bqv[8];
#pragma unroll
    for (int nt = 0; nt < 8; ++nt) {
        int n   = nt * 16 + ncol;
        bsum[nt] = bn[n] + bg[n];
        bkv[nt]  = bk[n];
        bqv[nt]  = bq[n];
    }

    // ---- Step 3: h = x@Wn^T + bn + broadcast(xg@Wg^T + bg)  -> bufH (f16) ----
#pragma unroll
    for (int nt = 0; nt < 8; ++nt) {
        v16h bwn[4], bwg[4];
#pragma unroll
        for (int ks = 0; ks < 4; ++ks) {        // 16 b128 loads issue as one clause
            bwn[ks] = ld_b_w(Wn16, nt, ks, lane);
            bwg[ks] = ld_b_w(Wg16, nt, ks, lane);
        }
        v8f c0 = {}, c1 = {}, cg = {};
#pragma unroll
        for (int ks = 0; ks < 4; ++ks) {        // 3 independent accumulator chains
            c0 = wmma16(a0[ks], bwn[ks], c0);
            c1 = wmma16(a1[ks], bwn[ks], c1);
            cg = wmma16(a1[ks], bwg[ks], cg);
        }
        int n = nt * 16 + ncol;
        float lg = cg[4] + bsum[nt];            // VGPR4 = rows 20/28 = lp_graph[n], both halves
#pragma unroll
        for (int r = 0; r < 8; ++r) {
            bufH[(r + rhi) * LROW + n]      = (_Float16)(c0[r] + lg);
            bufH[(16 + r + rhi) * LROW + n] = (_Float16)(c1[r] + lg);
        }
    }

    // ---- Step 4: load h operand tiles, then K and Q GEMMs fused.
    //      bufH is dead after ah[] is in registers -> Q overwrites bufH. ----
    v16h ah[2][4];
#pragma unroll
    for (int mt = 0; mt < 2; ++mt)
#pragma unroll
        for (int ks = 0; ks < 4; ++ks) ah[mt][ks] = ld_a_lds(bufH, mt, ks, lane);

    _Float16* bufQ = bufH;                      // reuse (DS ops of a wave are in-order)

#pragma unroll
    for (int nt = 0; nt < 8; ++nt) {
        v16h bwk[4], bwq[4];
#pragma unroll
        for (int ks = 0; ks < 4; ++ks) {
            bwk[ks] = ld_b_w(Wk16, nt, ks, lane);
            bwq[ks] = ld_b_w(Wq16, nt, ks, lane);
        }
        v8f k0 = {}, k1 = {}, q0 = {}, q1 = {}; // 4 independent accumulator chains
#pragma unroll
        for (int ks = 0; ks < 4; ++ks) {
            k0 = wmma16(ah[0][ks], bwk[ks], k0);
            k1 = wmma16(ah[1][ks], bwk[ks], k1);
            q0 = wmma16(ah[0][ks], bwq[ks], q0);
            q1 = wmma16(ah[1][ks], bwq[ks], q1);
        }
        int n = nt * 16 + ncol;
        float bbk = bkv[nt], bbq = bqv[nt];
#pragma unroll
        for (int r = 0; r < 8; ++r) {
            bufK[(r + rhi) * LROW + n]      = (_Float16)(k0[r] + bbk);
            bufK[(16 + r + rhi) * LROW + n] = (_Float16)(k1[r] + bbk);
            bufQ[(r + rhi) * LROW + n]      = (_Float16)(q0[r] + bbq);
            bufQ[(16 + r + rhi) * LROW + n] = (_Float16)(q1[r] + bbq);
        }
    }

    // ---- Step 5: M[n][m] = sum_d K[n,d]*Q[m,d], 2x2 tiles of 16x16 ----
    v16h ak[2][4];
#pragma unroll
    for (int tn = 0; tn < 2; ++tn)
#pragma unroll
        for (int ks = 0; ks < 4; ++ks) ak[tn][ks] = ld_a_lds(bufK, tn, ks, lane);

    v16h bqt[2][4];
#pragma unroll
    for (int tm = 0; tm < 2; ++tm)
#pragma unroll
        for (int ks = 0; ks < 4; ++ks) bqt[tm][ks] = ld_b_lds(bufQ, tm, ks, lane);

    v8f macc[2][2];
#pragma unroll
    for (int tn = 0; tn < 2; ++tn)
#pragma unroll
        for (int tm = 0; tm < 2; ++tm) macc[tn][tm] = (v8f){};

#pragma unroll
    for (int ks = 0; ks < 4; ++ks)              // 4 independent accumulator chains
#pragma unroll
        for (int tm = 0; tm < 2; ++tm) {
            macc[0][tm] = wmma16(ak[0][ks], bqt[tm][ks], macc[0][tm]);
            macc[1][tm] = wmma16(ak[1][ks], bqt[tm][ks], macc[1][tm]);
        }

    // ---- Step 6: masked tanh + flat-400 softmax + store ----
    const float scale = 0.08838834764831845f;   // 1/sqrt(128)
    float psum = 0.f;
#pragma unroll
    for (int tn = 0; tn < 2; ++tn)
#pragma unroll
        for (int tm = 0; tm < 2; ++tm)
#pragma unroll
            for (int r = 0; r < 8; ++r) {
                int nidx = tn * 16 + r + rhi;
                int midx = tm * 16 + ncol;
                float v  = macc[tn][tm][r] * scale;
                float t  = tanhf(v);
                float e  = 0.f;
                if (nidx < NNODE && midx < NNODE && nidx != midx)
                    e = __expf(10.f * t - 10.f);    // stable: 10*tanh <= 10
                macc[tn][tm][r] = e;                // diagonal -> exact 0
                psum += e;
            }
#pragma unroll
    for (int off = 16; off >= 1; off >>= 1) psum += __shfl_xor(psum, off, 32);
    const float inv = 1.f / psum;

    float* ob = out + (size_t)item * (NNODE * NNODE);
#pragma unroll
    for (int tn = 0; tn < 2; ++tn)
#pragma unroll
        for (int tm = 0; tm < 2; ++tm)
#pragma unroll
            for (int r = 0; r < 8; ++r) {
                int nidx = tn * 16 + r + rhi;
                int midx = tm * 16 + ncol;
                if (nidx < NNODE && midx < NNODE)
                    ob[nidx * NNODE + midx] = macc[tn][tm][r] * inv;
            }
}

// ------------------------------- launcher ----------------------------------
extern "C" void kernel_launch(void* const* d_in, const int* in_sizes, int n_in,
                              void* d_out, int out_size, void* d_ws, size_t ws_size,
                              hipStream_t stream) {
    // setup_inputs order: x, Wg, bg, Wn, bn, Wk, bk, Wq, bq (all f32)
    const float* x   = (const float*)d_in[0];
    const float* Wg  = (const float*)d_in[1];
    const float* bgp = (const float*)d_in[2];
    const float* Wn  = (const float*)d_in[3];
    const float* bnp = (const float*)d_in[4];
    const float* Wk  = (const float*)d_in[5];
    const float* bkp = (const float*)d_in[6];
    const float* Wq  = (const float*)d_in[7];
    const float* bqp = (const float*)d_in[8];

    _Float16* w16 = (_Float16*)d_ws;            // 4 * 128*128 halves = 128 KB
    cvt_weights_kernel<<<(DM * DM + 255) / 256, 256, 0, stream>>>(Wg, Wn, Wk, Wq, w16);

    const size_t lds_bytes = (size_t)WAVES * 2 * TILEH * sizeof(_Float16);  // 139264 B
    decoder_actor_kernel<<<BATCH / WAVES, WAVES * 32, lds_bytes, stream>>>(
        x, bgp, bnp, bkp, bqp, w16, (float*)d_out);
}
